// CombinedLoss_75170517615228
// MI455X (gfx1250) — matrix operand order repaired and checked by reading, not compiled, not run on hardware
//
#include <hip/hip_runtime.h>
#include <hip/hip_bf16.h>
#include <cstdint>

// ---------------- problem constants ----------------
#define BATCH 8
#define HDIM  384
#define WDIM  384
#define HW    (HDIM*WDIM)      // 147456
#define TOTN  (BATCH*HW)       // 1179648
#define NPAD  262144           // 2^18, sort padding per image
#define STILE 4096             // LDS-resident bitonic tile (32KB of u64)
#define INF2  294913.0f        // H*H + W*W + 1

typedef __attribute__((ext_vector_type(16))) _Float16 v16h;
typedef __attribute__((ext_vector_type(8)))  float    v8f;
typedef __attribute__((ext_vector_type(4)))  unsigned int v4u;
typedef __attribute__((ext_vector_type(8)))  int      v8i;
typedef __attribute__((ext_vector_type(4)))  int      v4i;
typedef unsigned long long u64;

#if defined(__has_builtin)
#  if __has_builtin(__builtin_amdgcn_tensor_load_to_lds) && __has_builtin(__builtin_amdgcn_s_wait_tensorcnt)
#    define USE_TDM 1
#  endif
#endif

// workspace float-index layout (see kernel_launch): partial region
#define PF_BASE  7471104            // = (dist TOTN) + (sort 16MB) + (cumsum 8MB) in floats
#define PF_MAX   (PF_BASE + 0)      // 1024 max-|dist| partials
#define PF_SCALE (PF_BASE + 1024)   // 1 scalar: 1/maxabs (or 1)
#define PF_GTS   (PF_BASE + 1040)   // 8 per-image sum(target)
#define PF_BSUM  (PF_BASE + 2048)   // 1024 scan block sums (128 per image)
#define PF_LOV   (PF_BASE + 4096)   // 1024 lovasz block partials
#define PF_SUMS  (PF_BASE + 8192)   // 4*1024 main partial sums

__device__ __forceinline__ float blockReduceSum256(float v, float* sh) {
  int t = threadIdx.x;
  sh[t] = v; __syncthreads();
  for (int s = 128; s > 0; s >>= 1) { if (t < s) sh[t] += sh[t + s]; __syncthreads(); }
  float r = sh[0]; __syncthreads();
  return r;
}

// ---------- K0: exact 1D EDT along rows (column-distance pass), O(W) ----------
__global__ void edt_colpass(const float* __restrict__ target,
                            float* __restrict__ g2pos, float* __restrict__ g2neg) {
  int row = blockIdx.x * blockDim.x + threadIdx.x;   // 0 .. BATCH*HDIM-1
  if (row >= BATCH * HDIM) return;
  const float* t = target + (size_t)row * WDIM;
  float* gp = g2pos + (size_t)row * WDIM;
  float* gn = g2neg + (size_t)row * WDIM;
  int lastF = -1, lastB = -1;
  for (int j = 0; j < WDIM; ++j) {
    bool fg = t[j] > 0.5f;
    if (fg) lastF = j; else lastB = j;
    float dF = (float)(j - lastF), dB = (float)(j - lastB);
    gp[j] = (lastF < 0) ? INF2 : dF * dF;
    gn[j] = (lastB < 0) ? INF2 : dB * dB;
  }
  int nextF = 1 << 20, nextB = 1 << 20;
  for (int j = WDIM - 1; j >= 0; --j) {
    bool fg = t[j] > 0.5f;
    if (fg) nextF = j; else nextB = j;
    if (nextF < (1 << 20)) { float d = (float)(nextF - j); gp[j] = fminf(gp[j], d * d); }
    if (nextB < (1 << 20)) { float d = (float)(nextB - j); gn[j] = fminf(gn[j], d * d); }
  }
}

// ---------- K1: row pass d2[i,j] = min_i' (i-i')^2 + g2[i',j], LDS-tiled, in-place ----
// Tile staging uses the Tensor Data Mover when available: one 2D D# descriptor
// (384 rows x 32 floats, row stride 384) DMA'd straight into LDS, tracked by
// TENSORcnt; falls back to plain vector loads otherwise.
__global__ void edt_rowpass(float* __restrict__ g2pos, float* __restrict__ g2neg) {
  __shared__ float tile[HDIM * 32];                  // 48 KB
  float* g2 = (blockIdx.z == 0) ? g2pos : g2neg;
  int img = blockIdx.y;
  int j0 = blockIdx.x * 32;
#if USE_TDM
  if (threadIdx.x == 0) {
    unsigned lds_off = (unsigned)(uintptr_t)&tile[0];          // low 32b of generic = LDS offset
    u64 ga = (u64)(uintptr_t)(g2 + (size_t)img * HW + j0);     // tile origin
    v4u g0;
    g0[0] = 1u;                                                // count=1, user descriptor
    g0[1] = lds_off;                                           // lds_addr (bytes)
    g0[2] = (unsigned)(ga & 0xffffffffull);                    // global_addr[31:0]
    g0[3] = (unsigned)((ga >> 32) & 0x01ffffffull) | (2u << 30); // addr[56:32] | type=2
    v8i g1;
    g1[0] = (int)(2u << 16);                                   // data_size=4B, no pad/multicast
    g1[1] = (int)(((unsigned)WDIM & 0xffffu) << 16);           // tensor_dim0 lo16
    g1[2] = (int)(((unsigned)WDIM >> 16) | (((unsigned)HDIM & 0xffffu) << 16)); // dim0 hi | dim1 lo
    g1[3] = (int)(((unsigned)HDIM >> 16) | (32u << 16));       // dim1 hi | tile_dim0=32
    g1[4] = (int)((unsigned)HDIM & 0xffffu);                   // tile_dim1=384, tile_dim2=0
    g1[5] = (int)WDIM;                                         // tensor_dim0_stride=384
    g1[6] = 0;
    g1[7] = 0;
    v4i z4 = {0, 0, 0, 0};
#  if __clang_major__ >= 23
    v8i z8 = {0, 0, 0, 0, 0, 0, 0, 0};
    __builtin_amdgcn_tensor_load_to_lds(g0, g1, z4, z4, z8, 0);
#  else
    __builtin_amdgcn_tensor_load_to_lds(g0, g1, z4, z4, 0);
#  endif
    __builtin_amdgcn_s_wait_tensorcnt(0);
  }
  __syncthreads();
#else
  for (int idx = threadIdx.x; idx < HDIM * 32; idx += 256) {
    int i = idx >> 5, j = idx & 31;
    tile[idx] = g2[((size_t)img * HDIM + i) * WDIM + j0 + j];
  }
  __syncthreads();
#endif
  for (int idx = threadIdx.x; idx < HDIM * 32; idx += 256) {
    int i = idx >> 5, j = idx & 31;
    float m = 1.0e30f;
    #pragma unroll 4
    for (int ip = 0; ip < HDIM; ++ip) {
      float di = (float)(i - ip);
      m = fminf(m, fmaf(di, di, tile[ip * 32 + j]));
    }
    g2[((size_t)img * HDIM + i) * WDIM + j0 + j] = m;
  }
}

// ---------- K2: signed distance map + block max|dist| ----------
__global__ void dist_and_max(const float* __restrict__ d2p, const float* __restrict__ d2n,
                             float* __restrict__ dist, float* __restrict__ maxP) {
  __shared__ float sh[256];
  float m = 0.0f;
  for (int i = blockIdx.x * blockDim.x + threadIdx.x; i < TOTN; i += gridDim.x * blockDim.x) {
    float v = sqrtf(fminf(d2p[i], INF2)) - sqrtf(fminf(d2n[i], INF2));
    dist[i] = v;
    m = fmaxf(m, fabsf(v));
  }
  sh[threadIdx.x] = m; __syncthreads();
  for (int s = 128; s > 0; s >>= 1) {
    if (threadIdx.x < s) sh[threadIdx.x] = fmaxf(sh[threadIdx.x], sh[threadIdx.x + s]);
    __syncthreads();
  }
  if (threadIdx.x == 0) maxP[blockIdx.x] = sh[0];
}

// ---------- K2b: per-image sum(target) = gts ----------
__global__ void image_sums(const float* __restrict__ target, float* __restrict__ gts) {
  __shared__ float sh[256];
  int img = blockIdx.x;
  float a = 0.0f;
  for (int i = threadIdx.x; i < HW; i += 256) a += target[(size_t)img * HW + i];
  float r = blockReduceSum256(a, sh);
  if (threadIdx.x == 0) gts[img] = r;
}

// ---------- K3: finalize max -> scale = maxabs>0 ? 1/maxabs : 1 ----------
__global__ void finalize_max(const float* __restrict__ maxP, float* __restrict__ scaleOut) {
  __shared__ float sh[256];
  float m = 0.0f;
  for (int i = threadIdx.x; i < 1024; i += 256) m = fmaxf(m, maxP[i]);
  sh[threadIdx.x] = m; __syncthreads();
  for (int s = 128; s > 0; s >>= 1) {
    if (threadIdx.x < s) sh[threadIdx.x] = fmaxf(sh[threadIdx.x], sh[threadIdx.x + s]);
    __syncthreads();
  }
  if (threadIdx.x == 0) scaleOut[0] = (sh[0] > 0.0f) ? (1.0f / sh[0]) : 1.0f;
}

// ---------- K4: build sortable 64-bit keys (ascending key == descending error) ------
__global__ void build_keys(const float* __restrict__ pred, const float* __restrict__ target,
                           u64* __restrict__ sortb) {
  int img = blockIdx.y;
  int idx = blockIdx.x * 256 + threadIdx.x;          // 0..NPAD-1
  u64 key;
  if (idx < HW) {
    float x = pred[(size_t)img * HW + idx];
    float t = target[(size_t)img * HW + idx];
    float e = 1.0f - x * (2.0f * t - 1.0f);          // hinge error
    unsigned u = __float_as_uint(e);
    unsigned s = (u & 0x80000000u) ? ~u : (u | 0x80000000u);  // order-preserving
    key = ((u64)(~s) << 32) | (u64)(t > 0.5f ? 1u : 0u);      // invert -> descending error
  } else {
    key = 0xFFFFFFFFFFFFFFFFull;                     // pads sort to the end
  }
  sortb[(size_t)img * NPAD + idx] = key;
}

// ---------- K5a: fused LDS bitonic, all stages k=2..STILE ----------
__global__ void bitonic_local_sort(u64* __restrict__ data) {
  __shared__ u64 sh[STILE];                          // 32 KB
  int img = blockIdx.y;
  int base = blockIdx.x * STILE;
  u64* d = data + (size_t)img * NPAD + base;
  for (int idx = threadIdx.x; idx < STILE; idx += 256) sh[idx] = d[idx];
  __syncthreads();
  for (int k = 2; k <= STILE; k <<= 1) {
    for (int j = k >> 1; j > 0; j >>= 1) {
      for (int idx = threadIdx.x; idx < STILE; idx += 256) {
        int ixj = idx ^ j;
        if (ixj > idx) {
          u64 a = sh[idx], b = sh[ixj];
          bool up = (((base + idx) & k) == 0);
          if ((a > b) == up) { sh[idx] = b; sh[ixj] = a; }
        }
      }
      __syncthreads();
    }
  }
  for (int idx = threadIdx.x; idx < STILE; idx += 256) d[idx] = sh[idx];
}

// ---------- K5b: one global bitonic compare-exchange pass (j >= STILE) ----------
__global__ void bitonic_step(u64* __restrict__ data, int j, int k) {
  int img = blockIdx.y;
  int i = blockIdx.x * 256 + threadIdx.x;
  u64* d = data + (size_t)img * NPAD;
  int ixj = i ^ j;
  if (ixj > i) {
    u64 a = d[i], b = d[ixj];
    bool up = ((i & k) == 0);
    if ((a > b) == up) { d[i] = b; d[ixj] = a; }
  }
}

// ---------- K5c: fused LDS finish of stage k (strides j = STILE/2 .. 1) ----------
__global__ void bitonic_local_finish(u64* __restrict__ data, int k) {
  __shared__ u64 sh[STILE];
  int img = blockIdx.y;
  int base = blockIdx.x * STILE;
  u64* d = data + (size_t)img * NPAD + base;
  for (int idx = threadIdx.x; idx < STILE; idx += 256) sh[idx] = d[idx];
  __syncthreads();
  for (int j = STILE >> 1; j > 0; j >>= 1) {
    for (int idx = threadIdx.x; idx < STILE; idx += 256) {
      int ixj = idx ^ j;
      if (ixj > idx) {
        u64 a = sh[idx], b = sh[ixj];
        bool up = (((base + idx) & k) == 0);
        if ((a > b) == up) { sh[idx] = b; sh[ixj] = a; }
      }
    }
    __syncthreads();
  }
  for (int idx = threadIdx.x; idx < STILE; idx += 256) d[idx] = sh[idx];
}

// ---------- K6: block-level inclusive scan of sorted gt bits via WMMA ----------
// Y = L(16x16 lower-tri ones) x X(16x16 segments) as one v_wmma_f32_16x16x32_f16
// (K zero-padded 16->32). f16 holds 0/1 exactly; f32 accumulation is bit-exact.
__global__ void scan_blocks_wmma(const u64* __restrict__ sortb,
                                 float* __restrict__ cums, float* __restrict__ bsum) {
  __shared__ float segT[8][16];
  __shared__ float waveTot[8];
  const int img  = blockIdx.y;
  const int blk  = blockIdx.x;                        // 0..127 (2048 elems per block)
  const int wave = threadIdx.x >> 5;                  // 0..7, 256 elems per wave
  const int lane = threadIdx.x & 31;
  const int mn   = lane & 15;                         // A: row M ; B/D: column N
  const int kbase = (lane >> 4) * 8;                  // A K-group / D M-group
  const u64* chunk = sortb + (size_t)img * NPAD + (size_t)blk * 2048 + (size_t)wave * 256;

  v16h a;
  #pragma unroll
  for (int h = 0; h < 16; ++h) {
    int K = 2 * kbase + h;
    a[h] = (_Float16)((K < 16 && K <= mn) ? 1.0f : 0.0f);
  }
  v16h b;
  if (lane < 16) {
    #pragma unroll
    for (int h = 0; h < 16; ++h)
      b[h] = (_Float16)(float)(unsigned)(chunk[mn * 16 + h] & 1ull);
  } else {
    #pragma unroll
    for (int h = 0; h < 16; ++h) b[h] = (_Float16)0.0f;
  }
  v8f c = {};
  c = __builtin_amdgcn_wmma_f32_16x16x32_f16(false, a, false, b, (short)0, c, false, false);

  if (lane >= 16) segT[wave][mn] = c[7];              // segment totals: D[15][n]
  __syncthreads();
  float segOff = 0.0f, wTot = 0.0f;
  #pragma unroll
  for (int q = 0; q < 16; ++q) {
    float v = segT[wave][q];
    segOff += (q < mn) ? v : 0.0f;
    wTot   += v;
  }
  if (lane == 0) waveTot[wave] = wTot;
  __syncthreads();
  float wOff = 0.0f;
  #pragma unroll
  for (int w = 0; w < 8; ++w) wOff += (w < wave) ? waveTot[w] : 0.0f;

  const float add = segOff + wOff;
  size_t outb = (size_t)img * NPAD + (size_t)blk * 2048 + (size_t)wave * 256
              + (size_t)mn * 16 + kbase;
  #pragma unroll
  for (int v = 0; v < 8; ++v) cums[outb + v] = c[v] + add;

  if (threadIdx.x == 0) {
    float bt = 0.0f;
    #pragma unroll
    for (int w = 0; w < 8; ++w) bt += waveTot[w];
    bsum[img * 128 + blk] = bt;
  }
}

// ---------- K7: exclusive scan of the 128 block sums per image ----------
__global__ void scan_block_sums(float* __restrict__ bsum) {
  __shared__ float s[128];
  int img = blockIdx.x, t = threadIdx.x;
  s[t] = bsum[img * 128 + t];
  __syncthreads();
  if (t == 0) {
    float acc = 0.0f;
    for (int q = 0; q < 128; ++q) { float v = s[q]; s[q] = acc; acc += v; }
  }
  __syncthreads();
  bsum[img * 128 + t] = s[t];
}

// ---------- K8: Lovasz contributions: relu(es_i) * grad_i, block-reduced ----------
__global__ void lovasz_contrib(const u64* __restrict__ sortb, const float* __restrict__ cums,
                               const float* __restrict__ bsum, const float* __restrict__ gtsArr,
                               float* __restrict__ lovP) {
  __shared__ float sh[256];
  int img = blockIdx.y, blk = blockIdx.x;
  float gts  = gtsArr[img];
  float boff = bsum[img * 128 + blk];
  float acc = 0.0f;
  for (int s = 0; s < 8; ++s) {
    int i = blk * 2048 + s * 256 + threadIdx.x;
    if (i < HW) {                                     // pads all land at positions >= HW
      u64 key = sortb[(size_t)img * NPAD + i];
      float Ci = cums[(size_t)img * NPAD + i] + boff; // inclusive cumsum of gt_sorted
      float g  = (float)(unsigned)(key & 1ull);
      unsigned sb = ~(unsigned)(key >> 32);
      unsigned u  = (sb & 0x80000000u) ? (sb & 0x7FFFFFFFu) : ~sb;
      float es = __uint_as_float(u);                  // sorted error value
      float fi = (float)i;
      float ji = 1.0f - (gts - Ci) / (gts + (fi + 1.0f) - Ci + 1e-7f);
      float grad;
      if (i == 0) grad = ji;
      else {
        float Cm = Ci - g;
        float jm = 1.0f - (gts - Cm) / (gts + fi - Cm + 1e-7f);
        grad = ji - jm;
      }
      acc += fmaxf(es, 0.0f) * grad;
    }
  }
  float r = blockReduceSum256(acc, sh);
  if (threadIdx.x == 0) lovP[img * 128 + blk] = r;
}

// ---------- K9: BCE / sigmoid / dice / boundary partial sums ----------
__global__ void main_sums(const float* __restrict__ pred, const float* __restrict__ target,
                          const float* __restrict__ dist, float* __restrict__ sums) {
  __shared__ float sh[256];
  float a0 = 0, a1 = 0, a2 = 0, a3 = 0;
  for (int i = blockIdx.x * blockDim.x + threadIdx.x; i < TOTN; i += gridDim.x * blockDim.x) {
    float x = pred[i], t = target[i];
    a0 += fmaxf(x, 0.0f) - x * t + log1pf(expf(-fabsf(x)));
    float p = 1.0f / (1.0f + expf(-x));
    a1 += p;
    a2 += p * t;
    a3 += p * dist[i];
  }
  float r;
  r = blockReduceSum256(a0, sh); if (threadIdx.x == 0) sums[0 * 1024 + blockIdx.x] = r;
  r = blockReduceSum256(a1, sh); if (threadIdx.x == 0) sums[1 * 1024 + blockIdx.x] = r;
  r = blockReduceSum256(a2, sh); if (threadIdx.x == 0) sums[2 * 1024 + blockIdx.x] = r;
  r = blockReduceSum256(a3, sh); if (threadIdx.x == 0) sums[3 * 1024 + blockIdx.x] = r;
}

// ---------- K10: combine everything into the scalar loss ----------
__global__ void finalize_loss(const float* __restrict__ wsF, float* __restrict__ out) {
  __shared__ float sh[256];
  int t = threadIdx.x;
  float s0 = 0, s1 = 0, s2 = 0, s3 = 0, lv = 0;
  for (int i = t; i < 1024; i += 256) {
    s0 += wsF[PF_SUMS + 0 * 1024 + i];
    s1 += wsF[PF_SUMS + 1 * 1024 + i];
    s2 += wsF[PF_SUMS + 2 * 1024 + i];
    s3 += wsF[PF_SUMS + 3 * 1024 + i];
    lv += wsF[PF_LOV + i];
  }
  float Sbce = blockReduceSum256(s0, sh);
  float Sp   = blockReduceSum256(s1, sh);
  float Spt  = blockReduceSum256(s2, sh);
  float Spd  = blockReduceSum256(s3, sh);
  float Lsum = blockReduceSum256(lv, sh);
  if (t == 0) {
    float St = 0.0f;
    for (int b = 0; b < BATCH; ++b) St += wsF[PF_GTS + b];
    float scale = wsF[PF_SCALE];
    float bce      = Sbce / (float)TOTN;
    float dice     = 1.0f - (2.0f * Spt + 1.0f) / (Sp + St + 1.0f);
    float boundary = scale * Spd / (float)TOTN;
    float lovasz   = Lsum / (float)BATCH;
    float loss = 0.3f * bce + 0.3f * dice + 0.2f * boundary + 0.2f * lovasz;
    out[0] = fminf(fmaxf(loss, 0.0f), 100.0f);
  }
}

// ---------------- host launch ----------------
extern "C" void kernel_launch(void* const* d_in, const int* in_sizes, int n_in,
                              void* d_out, int out_size, void* d_ws, size_t ws_size,
                              hipStream_t stream) {
  const float* pred   = (const float*)d_in[0];
  const float* target = (const float*)d_in[1];
  float* out = (float*)d_out;

  // workspace layout (bytes):
  //   [0, 4.72MB)        dist map (f32 TOTN)
  //   [4.72MB, 21.5MB)   sort buffer (u64 BATCH*NPAD); EDT temporaries g2pos/g2neg
  //                      alias this region (dead before the sort starts)
  //   [21.5MB, 29.9MB)   cumsum buffer (f32 BATCH*NPAD)
  //   [29.9MB, +64KB)    partial/scalar region (PF_* float offsets)
  char* ws = (char*)d_ws;
  float* dist  = (float*)ws;
  u64*   sortb = (u64*)(ws + (size_t)TOTN * 4);
  float* g2pos = (float*)(ws + (size_t)TOTN * 4);
  float* g2neg = g2pos + TOTN;
  float* cums  = (float*)(ws + (size_t)TOTN * 4 + (size_t)BATCH * NPAD * 8);
  float* wsF   = (float*)ws;

  // 1) exact EDT (signed distance map): O(N) col pass + TDM/LDS-tiled row pass
  edt_colpass<<<dim3((BATCH * HDIM + 255) / 256), dim3(256), 0, stream>>>(target, g2pos, g2neg);
  edt_rowpass<<<dim3(WDIM / 32, BATCH, 2), dim3(256), 0, stream>>>(g2pos, g2neg);
  dist_and_max<<<dim3(1024), dim3(256), 0, stream>>>(g2pos, g2neg, dist, wsF + PF_MAX);
  image_sums<<<dim3(BATCH), dim3(256), 0, stream>>>(target, wsF + PF_GTS);
  finalize_max<<<dim3(1), dim3(256), 0, stream>>>(wsF + PF_MAX, wsF + PF_SCALE);

  // 2) Lovasz: key build + hybrid LDS/global bitonic sort (28 launches, mostly LDS)
  build_keys<<<dim3(NPAD / 256, BATCH), dim3(256), 0, stream>>>(pred, target, sortb);
  bitonic_local_sort<<<dim3(NPAD / STILE, BATCH), dim3(256), 0, stream>>>(sortb);
  for (int k = STILE * 2; k <= NPAD; k <<= 1) {
    for (int j = k >> 1; j >= STILE; j >>= 1)
      bitonic_step<<<dim3(NPAD / 256, BATCH), dim3(256), 0, stream>>>(sortb, j, k);
    bitonic_local_finish<<<dim3(NPAD / STILE, BATCH), dim3(256), 0, stream>>>(sortb, k);
  }

  // 3) cumsum of sorted labels via WMMA triangular matmul, then carries
  scan_blocks_wmma<<<dim3(NPAD / 2048, BATCH), dim3(256), 0, stream>>>(sortb, cums, wsF + PF_BSUM);
  scan_block_sums<<<dim3(BATCH), dim3(128), 0, stream>>>(wsF + PF_BSUM);
  lovasz_contrib<<<dim3(NPAD / 2048, BATCH), dim3(256), 0, stream>>>(
      sortb, cums, wsF + PF_BSUM, wsF + PF_GTS, wsF + PF_LOV);

  // 4) elementwise terms + deterministic tree reductions, final combine
  main_sums<<<dim3(1024), dim3(256), 0, stream>>>(pred, target, dist, wsF + PF_SUMS);
  finalize_loss<<<dim3(1), dim3(256), 0, stream>>>(wsF, out);
}
